// Attn_17343077941439
// MI455X (gfx1250) — matrix operand order repaired and checked by reading, not compiled, not run on hardware
//
#include <hip/hip_runtime.h>
#include <hip/hip_bf16.h>
#include <cstdint>
#include <cstddef>

#define B_  2
#define S_  2048
#define H_  16
#define DH  64
#define DM  1024
#define BS  (B_ * S_)   // 4096

typedef __attribute__((ext_vector_type(16))) __bf16   v16bf;
typedef __attribute__((ext_vector_type(8)))  float    v8f;
typedef __attribute__((ext_vector_type(4)))  unsigned u32x4;
typedef __attribute__((ext_vector_type(4)))  float    f32x4;

struct U8 { u32x4 lo, hi; };

__device__ __forceinline__ __bf16 f2bf(float f) {
  unsigned u = __builtin_bit_cast(unsigned, f);
  u += 0x7FFFu + ((u >> 16) & 1u);            // round-to-nearest-even
  unsigned short h = (unsigned short)(u >> 16);
  return __builtin_bit_cast(__bf16, h);
}

// Build a v16bf WMMA operand from two 16-byte aligned runs of 8 bf16 each.
__device__ __forceinline__ v16bf make_frag(const __bf16* p0, const __bf16* p1) {
  U8 u;
  u.lo = *(const u32x4*)p0;
  u.hi = *(const u32x4*)p1;
  return __builtin_bit_cast(v16bf, u);
}

__device__ __forceinline__ v8f wmma_bf16(v16bf a, v16bf b, v8f c) {
  return __builtin_amdgcn_wmma_f32_16x16x32_bf16(false, a, false, b, (short)0, c,
                                                 false, false);
}

// A-fragment: 16x32 bf16, lane = row (lane&15); K runs {0..15} / {16..31}
__device__ __forceinline__ v16bf ldA(const __bf16* row, int k, int half) {
  const __bf16* p = row + k + half * 16;
  return make_frag(p, p + 8);
}
// B-fragment: 32x16 bf16, lane = col; 16 contiguous K values per lane
__device__ __forceinline__ v16bf ldB(const __bf16* row, int k) {
  return make_frag(row + k, row + k + 8);
}

// 32(M) x 64(N) x 1024(K) register-pipelined GEMM core, two ping-pong stages,
// no cross-stage register copies.
__device__ __forceinline__ void gemm32x64(const __bf16* ar0, const __bf16* ar1,
                                          const __bf16* br0, const __bf16* br1,
                                          const __bf16* br2, const __bf16* br3,
                                          int half, v8f acc[2][4]) {
  const __bf16* ar[2] = {ar0, ar1};
  const __bf16* br[4] = {br0, br1, br2, br3};
  v16bf a0[2], a1[2], b0[4], b1[4];
#pragma unroll
  for (int i = 0; i < 2; ++i) { a0[i] = ldA(ar[i], 0, half); a1[i] = ldA(ar[i], 32, half); }
#pragma unroll
  for (int t = 0; t < 4; ++t) { b0[t] = ldB(br[t], 0); b1[t] = ldB(br[t], 32); }

  for (int k = 0; k < DM; k += 64) {
    // ---- stage 0: compute k, then refill stage 0 for k+64
#pragma unroll
    for (int i = 0; i < 2; ++i)
#pragma unroll
      for (int t = 0; t < 4; ++t) acc[i][t] = wmma_bf16(a0[i], b0[t], acc[i][t]);
    int k0 = (k + 64 < DM) ? k + 64 : 0;          // clamped dummy reload on last
#pragma unroll
    for (int i = 0; i < 2; ++i) a0[i] = ldA(ar[i], k0, half);
#pragma unroll
    for (int t = 0; t < 4; ++t) b0[t] = ldB(br[t], k0);

    // ---- stage 1: compute k+32, then refill stage 1 for k+96
#pragma unroll
    for (int i = 0; i < 2; ++i)
#pragma unroll
      for (int t = 0; t < 4; ++t) acc[i][t] = wmma_bf16(a1[i], b1[t], acc[i][t]);
    int k1 = (k + 96 < DM) ? k + 96 : 32;
#pragma unroll
    for (int i = 0; i < 2; ++i) a1[i] = ldA(ar[i], k1, half);
#pragma unroll
    for (int t = 0; t < 4; ++t) b1[t] = ldB(br[t], k1);
  }
}

// ---------------------------------------------------------------- convert x
__global__ void k_cvt(const float* __restrict__ src, __bf16* __restrict__ dst, int n) {
  int i = (blockIdx.x * blockDim.x + threadIdx.x) * 4;
  if (i + 3 < n) {
    f32x4 v = *(const f32x4*)(src + i);
    dst[i + 0] = f2bf(v.x);
    dst[i + 1] = f2bf(v.y);
    dst[i + 2] = f2bf(v.z);
    dst[i + 3] = f2bf(v.w);
  }
}

// ----------------------------------------------- transpose 1024x1024 weight
__global__ void k_tr(const float* __restrict__ w, __bf16* __restrict__ wt) {
  int n = blockIdx.x * 16 + threadIdx.x;   // output feature (coalesced read)
  int k = blockIdx.y * 16 + threadIdx.y;   // input feature
  wt[(size_t)n * DM + k] = f2bf(w[(size_t)k * DM + n]);
}

// -------------------------------------------------------- fused QKV GEMM
__global__ void __launch_bounds__(128) k_qkv(
    const __bf16* __restrict__ xh,
    const __bf16* __restrict__ wqT, const __bf16* __restrict__ wkT,
    const __bf16* __restrict__ wvT,
    const float* __restrict__ bq, const float* __restrict__ bk,
    const float* __restrict__ bv,
    __bf16* __restrict__ Qb, __bf16* __restrict__ Kb, __bf16* __restrict__ Vt) {
  const int lane = threadIdx.x & 31;
  const int wv   = threadIdx.x >> 5;
  const int task = blockIdx.x * 4 + wv;     // 6144 tasks = 128 * 16 * 3
  const int mt    = task & 127;
  const int rest  = task >> 7;
  const int ns    = rest & 15;
  const int which = rest >> 4;              // 0=Q 1=K 2=V

  const __bf16* wT   = which == 0 ? wqT : (which == 1 ? wkT : wvT);
  const float*  bias = which == 0 ? bq  : (which == 1 ? bk  : bv);

  const int half = lane >> 4;
  const int l16  = lane & 15;
  const int m0 = mt * 32;
  const int n0 = ns * 64;

  const __bf16* ar0 = xh + (size_t)(m0 + l16) * DM;
  const __bf16* ar1 = ar0 + (size_t)16 * DM;
  const __bf16* br0 = wT + (size_t)(n0 +  0 + l16) * DM + half * 16;
  const __bf16* br1 = wT + (size_t)(n0 + 16 + l16) * DM + half * 16;
  const __bf16* br2 = wT + (size_t)(n0 + 32 + l16) * DM + half * 16;
  const __bf16* br3 = wT + (size_t)(n0 + 48 + l16) * DM + half * 16;

  v8f acc[2][4];
#pragma unroll
  for (int i = 0; i < 2; ++i)
#pragma unroll
    for (int t = 0; t < 4; ++t) acc[i][t] = (v8f){};

  gemm32x64(ar0, ar1, br0, br1, br2, br3, half, acc);

#pragma unroll
  for (int i = 0; i < 2; ++i) {
#pragma unroll
    for (int r = 0; r < 8; ++r) {
      int sg = m0 + i * 16 + r + (half << 3);
      int bb = sg >> 11, s = sg & (S_ - 1);
#pragma unroll
      for (int t = 0; t < 4; ++t) {
        int f = n0 + t * 16 + l16;
        float v = acc[i][t][r] + bias[f];
        int h = f >> 6, d = f & 63;
        size_t bh = (size_t)bb * H_ + h;
        if (which == 0)       Qb[(bh * S_ + s) * DH + d] = f2bf(v * 0.125f);
        else if (which == 1)  Kb[(bh * S_ + s) * DH + d] = f2bf(v);
        else                  Vt[(bh * DH + d) * S_ + s] = f2bf(v);
      }
    }
  }
}

// ------------------------------------------------- flash attention per wave
__global__ void __launch_bounds__(128) k_flash(
    const __bf16* __restrict__ Qb, const __bf16* __restrict__ Kb,
    const __bf16* __restrict__ Vt, __bf16* __restrict__ attn) {
  __shared__ __attribute__((aligned(16))) __bf16 pP[4][16][40];  // 40: bank stagger

  const int lane = threadIdx.x & 31;
  const int wv   = threadIdx.x >> 5;
  const int task = blockIdx.x * 4 + wv;    // 4096 tasks = 32 * 128
  const int qt = task & 127;
  const int bh = task >> 7;
  const int q0 = qt * 16;
  const int half = lane >> 4, l16 = lane & 15;

  const __bf16* Qh = Qb + (size_t)bh * S_ * DH;
  const __bf16* Kh = Kb + (size_t)bh * S_ * DH;
  const __bf16* Vh = Vt + (size_t)bh * DH * S_;

  const __bf16* qrow = Qh + (size_t)(q0 + l16) * DH + half * 16;
  v16bf qa0 = make_frag(qrow,      qrow +  8);   // d 0..31
  v16bf qa1 = make_frag(qrow + 32, qrow + 40);   // d 32..63

  v8f o0 = {}, o1 = {}, o2 = {}, o3 = {};
  float m[8], l[8];
#pragma unroll
  for (int r = 0; r < 8; ++r) { m[r] = -1e30f; l[r] = 0.f; }

  // K fragments for key-block kb: 2 key-tiles x 2 d-chunks
  const __bf16* Kbase = Kh + (size_t)l16 * DH + half * 16;
  v16bf kf[4];
  {
    const __bf16* kr0 = Kbase;
    const __bf16* kr1 = Kbase + (size_t)16 * DH;
    kf[0] = make_frag(kr0,      kr0 +  8);
    kf[1] = make_frag(kr0 + 32, kr0 + 40);
    kf[2] = make_frag(kr1,      kr1 +  8);
    kf[3] = make_frag(kr1 + 32, kr1 + 40);
  }

  const int nblk = (q0 + 15) / 32 + 1;     // causal frontier, 32 keys per block
  for (int blk = 0; blk < nblk; ++blk) {
    int kb = blk * 32;

    v8f s1 = {}, s2 = {};
    s1 = wmma_bf16(qa0, kf[0], s1);
    s1 = wmma_bf16(qa1, kf[1], s1);
    s2 = wmma_bf16(qa0, kf[2], s2);
    s2 = wmma_bf16(qa1, kf[3], s2);

    // V fragments issued early: latency hidden under softmax VALU work.
    const __bf16* vr0 = Vh + (size_t)( 0 + l16) * S_ + kb + half * 16;
    const __bf16* vr1 = Vh + (size_t)(16 + l16) * S_ + kb + half * 16;
    const __bf16* vr2 = Vh + (size_t)(32 + l16) * S_ + kb + half * 16;
    const __bf16* vr3 = Vh + (size_t)(48 + l16) * S_ + kb + half * 16;
    v16bf vf0 = make_frag(vr0, vr0 + 8);
    v16bf vf1 = make_frag(vr1, vr1 + 8);
    v16bf vf2 = make_frag(vr2, vr2 + 8);
    v16bf vf3 = make_frag(vr3, vr3 + 8);

    // Prefetch next block's K fragments into the spare register set
    // (clamped dummy reload on the last block -> no divergence).
    int kbn = (blk + 1 < nblk) ? kb + 32 : kb;
    v16bf kn[4];
    {
      const __bf16* kr0 = Kbase + (size_t)kbn * DH;
      const __bf16* kr1 = Kbase + (size_t)(kbn + 16) * DH;
      kn[0] = make_frag(kr0,      kr0 +  8);
      kn[1] = make_frag(kr0 + 32, kr0 + 40);
      kn[2] = make_frag(kr1,      kr1 +  8);
      kn[3] = make_frag(kr1 + 32, kr1 + 40);
    }
    // Warm L2 for block+2 (global_prefetch_b8).
    if (kb + 64 < S_) {
      __builtin_prefetch(Kh + (size_t)(kb + 64 + lane) * DH, 0, 3);
      __builtin_prefetch(Vh + (size_t)(lane + half * 16) * S_ + kb + 64, 0, 3);
    }

    // causal mask + online softmax; C-layout row M = r + 8*half, N = l16
#pragma unroll
    for (int r = 0; r < 8; ++r) {
      int q = q0 + r + (half << 3);
      float v1 = s1[r]; if (kb +      l16 > q) v1 = -1e30f;
      float v2 = s2[r]; if (kb + 16 + l16 > q) v2 = -1e30f;
      float cm = fmaxf(v1, v2);
      cm = fmaxf(cm, __shfl_xor(cm, 1, 32));
      cm = fmaxf(cm, __shfl_xor(cm, 2, 32));
      cm = fmaxf(cm, __shfl_xor(cm, 4, 32));
      cm = fmaxf(cm, __shfl_xor(cm, 8, 32));
      float mn = fmaxf(m[r], cm);
      float sc = __expf(m[r] - mn);
      float p1 = __expf(v1 - mn);
      float p2 = __expf(v2 - mn);
      float ps = p1 + p2;
      ps += __shfl_xor(ps, 1, 32);
      ps += __shfl_xor(ps, 2, 32);
      ps += __shfl_xor(ps, 4, 32);
      ps += __shfl_xor(ps, 8, 32);
      l[r] = l[r] * sc + ps;
      m[r] = mn;
      o0[r] *= sc; o1[r] *= sc; o2[r] *= sc; o3[r] *= sc;
      int rowM = r + (half << 3);
      pP[wv][rowM][l16]      = f2bf(p1);
      pP[wv][rowM][l16 + 16] = f2bf(p2);
    }

    // cross-lane C->A relayout through LDS: wait for our wave's DS stores
    asm volatile("s_wait_dscnt 0x0" ::: "memory");
    const __bf16* prow = &pP[wv][l16][half * 16];
    v16bf pa = make_frag(prow, prow + 8);

    // PV: P (16x32) x V (32x64); V^T storage makes key-dim contiguous
    o0 = wmma_bf16(pa, vf0, o0);
    o1 = wmma_bf16(pa, vf1, o1);
    o2 = wmma_bf16(pa, vf2, o2);
    o3 = wmma_bf16(pa, vf3, o3);

    // rotate K stages (waits for kn land here, after softmax + PV)
#pragma unroll
    for (int t = 0; t < 4; ++t) kf[t] = kn[t];
  }

  const int b = bh >> 4, h = bh & 15;
#pragma unroll
  for (int r = 0; r < 8; ++r) {
    float inv = 1.0f / l[r];
    int s = q0 + r + (half << 3);
    size_t base = ((size_t)(b * S_ + s)) * DM + h * DH + l16;
    attn[base +  0] = f2bf(o0[r] * inv);
    attn[base + 16] = f2bf(o1[r] * inv);
    attn[base + 32] = f2bf(o2[r] * inv);
    attn[base + 48] = f2bf(o3[r] * inv);
  }
}

// ------------------------------------------------------- output projection
__global__ void __launch_bounds__(128) k_proj(
    const __bf16* __restrict__ ah, const __bf16* __restrict__ woT,
    float* __restrict__ out) {
  const int lane = threadIdx.x & 31;
  const int wv   = threadIdx.x >> 5;
  const int task = blockIdx.x * 4 + wv;   // 2048 tasks = 128 * 16
  const int mt = task & 127;
  const int ns = task >> 7;

  const int half = lane >> 4;
  const int l16  = lane & 15;
  const int m0 = mt * 32;
  const int n0 = ns * 64;

  const __bf16* ar0 = ah + (size_t)(m0 + l16) * DM;
  const __bf16* ar1 = ar0 + (size_t)16 * DM;
  const __bf16* br0 = woT + (size_t)(n0 +  0 + l16) * DM + half * 16;
  const __bf16* br1 = woT + (size_t)(n0 + 16 + l16) * DM + half * 16;
  const __bf16* br2 = woT + (size_t)(n0 + 32 + l16) * DM + half * 16;
  const __bf16* br3 = woT + (size_t)(n0 + 48 + l16) * DM + half * 16;

  v8f acc[2][4];
#pragma unroll
  for (int i = 0; i < 2; ++i)
#pragma unroll
    for (int t = 0; t < 4; ++t) acc[i][t] = (v8f){};

  gemm32x64(ar0, ar1, br0, br1, br2, br3, half, acc);

#pragma unroll
  for (int i = 0; i < 2; ++i) {
#pragma unroll
    for (int r = 0; r < 8; ++r) {
      int sg = m0 + i * 16 + r + (half << 3);
      float* orow = out + (size_t)sg * DM + n0 + l16;
      orow[ 0] = acc[i][0][r];
      orow[16] = acc[i][1][r];
      orow[32] = acc[i][2][r];
      orow[48] = acc[i][3][r];
    }
  }
}

extern "C" void kernel_launch(void* const* d_in, const int* in_sizes, int n_in,
                              void* d_out, int out_size, void* d_ws, size_t ws_size,
                              hipStream_t stream) {
  const float* x  = (const float*)d_in[0];
  const float* Wq = (const float*)d_in[1];
  const float* bq = (const float*)d_in[2];
  const float* Wk = (const float*)d_in[3];
  const float* bk = (const float*)d_in[4];
  const float* Wv = (const float*)d_in[5];
  const float* bv = (const float*)d_in[6];
  const float* Wo = (const float*)d_in[7];

  char* ws = (char*)d_ws;
  __bf16* xh  = (__bf16*)(ws + ((size_t) 0 << 20));  //  8 MiB  x in bf16
  __bf16* wqT = (__bf16*)(ws + ((size_t) 8 << 20));  //  2 MiB
  __bf16* wkT = (__bf16*)(ws + ((size_t)10 << 20));  //  2 MiB
  __bf16* wvT = (__bf16*)(ws + ((size_t)12 << 20));  //  2 MiB
  __bf16* woT = (__bf16*)(ws + ((size_t)14 << 20));  //  2 MiB
  __bf16* Qb  = (__bf16*)(ws + ((size_t)16 << 20));  //  8 MiB  [B,H,S,64] (pre-scaled)
  __bf16* Kb  = (__bf16*)(ws + ((size_t)24 << 20));  //  8 MiB  [B,H,S,64]
  __bf16* Vt  = (__bf16*)(ws + ((size_t)32 << 20));  //  8 MiB  [B,H,64,S]
  __bf16* ah  = (__bf16*)(ws + ((size_t)40 << 20));  //  8 MiB  [B,S,1024]

  k_cvt<<<4096, 256, 0, stream>>>(x, xh, BS * DM);
  dim3 tb(16, 16), tg(64, 64);
  k_tr<<<tg, tb, 0, stream>>>(Wq, wqT);
  k_tr<<<tg, tb, 0, stream>>>(Wk, wkT);
  k_tr<<<tg, tb, 0, stream>>>(Wv, wvT);
  k_tr<<<tg, tb, 0, stream>>>(Wo, woT);

  k_qkv<<<1536, 128, 0, stream>>>(xh, wqT, wkT, wvT, bq, bk, bv, Qb, Kb, Vt);
  k_flash<<<1024, 128, 0, stream>>>(Qb, Kb, Vt, ah);
  k_proj<<<512, 128, 0, stream>>>(ah, woT, (float*)d_out);
}
